// MaskLSTM_74002286510360
// MI455X (gfx1250) — matrix operand order repaired and checked by reading, not compiled, not run on hardware
//
#include <hip/hip_runtime.h>
#include <hip/hip_bf16.h>

#define B_   64
#define T_   1024
#define NIN  512
#define H_   512
#define G4   2048
#define WPAD 520   // 512 + 8 bf16 pad (16B), matches TDM pad interval=1024B
#define HPAD 520
#define APAD 72    // 64 + 8 bf16 pad (16B), matches TDM pad interval=128B

typedef __attribute__((ext_vector_type(16))) __bf16 v16bf;
typedef __attribute__((ext_vector_type(8)))  __bf16 v8bf;
typedef __attribute__((ext_vector_type(8)))  float  v8f;
typedef __attribute__((ext_vector_type(4)))  unsigned int v4u;
typedef __attribute__((ext_vector_type(8)))  int v8i;
typedef __attribute__((ext_vector_type(4)))  int v4i;

#if __has_builtin(__builtin_amdgcn_tensor_load_to_lds)
#define HAVE_TDM 1
#else
#define HAVE_TDM 0
#endif

#if HAVE_TDM
// 2-D bf16 tile load Global->LDS via Tensor Data Mover (CDNA5 ISA ch.8).
// pad_i/pad_a are the encoded pad_interval/pad_amount codes; pad_i<0 = no pad.
__device__ __forceinline__ void tdm_load_2d_bf16(
    unsigned lds_off, const void* gptr,
    unsigned tensor_d0, unsigned tensor_d1, unsigned long long stride0,
    unsigned tile_d0, unsigned tile_d1, int pad_i, int pad_a)
{
    unsigned long long ga = (unsigned long long)(uintptr_t)gptr;
    v4u g0;
    g0[0] = 1u;                                   // count=1, user descriptor
    g0[1] = lds_off;                              // LDS byte address
    g0[2] = (unsigned)(ga & 0xffffffffu);         // global_addr[31:0]
    g0[3] = (unsigned)((ga >> 32) & 0x01ffffffu)  // global_addr[56:32]
          | 0x80000000u;                          // type=2 (bits 127:126)
    unsigned w0 = (1u << 16);                     // data_size=1 -> 2 bytes
    if (pad_i >= 0)
        w0 |= (1u << 20) | ((unsigned)pad_i << 22) | ((unsigned)pad_a << 25);
    v8i g1;
    g1[0] = (int)w0;
    g1[1] = (int)((tensor_d0 & 0xffffu) << 16);                 // dim0[15:0]
    g1[2] = (int)(((tensor_d0 >> 16) & 0xffffu)
          |       ((tensor_d1 & 0xffffu) << 16));               // dim0[31:16]|dim1[15:0]
    g1[3] = (int)(((tensor_d1 >> 16) & 0xffffu)
          |       ((tile_d0 & 0xffffu) << 16));                 // dim1[31:16]|tile_dim0
    g1[4] = (int)(tile_d1 & 0xffffu);                           // tile_dim1 (dim2=0)
    g1[5] = (int)(unsigned)(stride0 & 0xffffffffu);             // dim0_stride[31:0]
    g1[6] = (int)(unsigned)((stride0 >> 32) & 0xffffu);         // dim0_stride[47:32]
    g1[7] = 0;
    v4i z4 = {0, 0, 0, 0};
#if __clang_major__ >= 23
    v8i z8 = {0, 0, 0, 0, 0, 0, 0, 0};
    __builtin_amdgcn_tensor_load_to_lds(g0, g1, z4, z4, z8, 0);
#else
    __builtin_amdgcn_tensor_load_to_lds(g0, g1, z4, z4, 0);
#endif
}

__device__ __forceinline__ void tdm_wait() {
#if __has_builtin(__builtin_amdgcn_s_wait_tensorcnt)
    __builtin_amdgcn_s_wait_tensorcnt(0);
#else
    asm volatile("s_wait_tensorcnt 0x0" ::: "memory");
#endif
}
#endif  // HAVE_TDM

// ---------------------------------------------------------------------------
// Pre-pass A: X (f32) -> Xb (bf16), 8 elements/thread
// ---------------------------------------------------------------------------
__global__ __launch_bounds__(256) void cvt_x_kernel(
    const float* __restrict__ X, __bf16* __restrict__ Xb)
{
    const size_t i = ((size_t)blockIdx.x * 256 + threadIdx.x) * 8;
    const float4 a = *(const float4*)(X + i);
    const float4 b = *(const float4*)(X + i + 4);
    v8bf o;
    o[0] = (__bf16)a.x; o[1] = (__bf16)a.y; o[2] = (__bf16)a.z; o[3] = (__bf16)a.w;
    o[4] = (__bf16)b.x; o[5] = (__bf16)b.y; o[6] = (__bf16)b.z; o[7] = (__bf16)b.w;
    *(v8bf*)(Xb + i) = o;
}

// ---------------------------------------------------------------------------
// Pre-pass B: W[k][g] (f32) -> WT[g][k] (bf16) for W_x and W_h (blockIdx.y)
// ---------------------------------------------------------------------------
__global__ __launch_bounds__(256) void transpose_w_kernel(
    const float* __restrict__ Wx, const float* __restrict__ Wh,
    __bf16* __restrict__ WxT, __bf16* __restrict__ WhT)
{
    const float* src = blockIdx.y ? Wh : Wx;
    __bf16*      dst = blockIdx.y ? WhT : WxT;
    const size_t i = (size_t)blockIdx.x * 256 + threadIdx.x;   // over 2048*512
    const int g = (int)(i >> 9);
    const int k = (int)(i & 511);
    dst[i] = (__bf16)src[(size_t)k * G4 + g];
}

// ---------------------------------------------------------------------------
// Phase 1: XP[t][b][g] (bf16) = Xb[b][t][:] . WxT[g][:] + bx[g]
// Grid (16 col-blocks, 16 t-blocks of 64, 64 batches), 256 thr (8 waves).
// WxT col-block TDM-staged in LDS once; X streamed as eight 64x64 chunks
// through a 2-deep TDM double buffer overlapped with WMMA.
// ---------------------------------------------------------------------------
__global__ __launch_bounds__(256) void xproj_kernel(
    const __bf16* __restrict__ Xb, const __bf16* __restrict__ WxT,
    const float* __restrict__ bx, __bf16* __restrict__ XP)
{
    extern __shared__ char smem_raw[];
    __bf16* Wt = (__bf16*)smem_raw;                          // [128][WPAD]
    __bf16* aL = (__bf16*)(smem_raw + 128 * WPAD * 2);       // [2][64][APAD]

    const int cb   = blockIdx.x;
    const int t0   = blockIdx.y * 64;
    const int b    = blockIdx.z;
    const int lane = threadIdx.x & 31;
    const int w    = threadIdx.x >> 5;               // n-tile 0..7
    const int col  = lane & 15;
    const int hsel = lane >> 4;
    const int g    = cb * 128 + w * 16 + col;
    const int koffA = hsel * 8;

    const __bf16* Arow0 = Xb + ((size_t)b * T_ + t0) * NIN;

#if HAVE_TDM
    if (threadIdx.x < 32) {
        tdm_load_2d_bf16((unsigned)(uintptr_t)Wt, WxT + (size_t)cb * 128 * 512,
                         512, 2048, 512, /*tile*/512, 128, /*pad 1024B/16B*/7, 3);
        tdm_load_2d_bf16((unsigned)(uintptr_t)aL, Arow0,
                         512, 1024, 512, /*tile*/64, 64, /*pad 128B/16B*/4, 3);
        tdm_wait();
    }
    __syncthreads();
#else
    for (int idx = threadIdx.x; idx < 128 * 512; idx += 256) {
        const int c2 = idx >> 9, k = idx & 511;
        Wt[c2 * WPAD + k] = WxT[((size_t)cb * 128 + c2) * 512 + k];
    }
#endif

    v8f acc0 = {0,0,0,0,0,0,0,0}, acc1 = acc0, acc2 = acc0, acc3 = acc0;

    for (int c = 0; c < 8; ++c) {                    // K chunks of 64
        __bf16* abuf = aL + (c & 1) * 64 * APAD;
#if HAVE_TDM
        if (threadIdx.x < 32 && c + 1 < 8)           // prefetch next chunk via TDM
            tdm_load_2d_bf16((unsigned)(uintptr_t)(aL + ((c + 1) & 1) * 64 * APAD),
                             Arow0 + 64 * (c + 1),
                             512, 1024, 512, 64, 64, 4, 3);
#else
        for (int ii = threadIdx.x; ii < 64 * 8; ii += 256) {
            const int rr = ii >> 3, cc8 = (ii & 7) * 8;
            *(v8bf*)&abuf[rr * APAD + cc8] =
                *(const v8bf*)&Arow0[(size_t)rr * NIN + 64 * c + cc8];
        }
        __syncthreads();
#endif
        #pragma unroll
        for (int kbl = 0; kbl < 64; kbl += 32) {
            const int kg = 64 * c + kbl;
            // B fragment from LDS-resident WxT tile
            const __bf16* wp = &Wt[(w * 16 + col) * WPAD + kg + hsel * 8];
            const v8bf b0 = *(const v8bf*)wp;
            const v8bf b1 = *(const v8bf*)(wp + 16);
            v16bf bf;
            #pragma unroll
            for (int i = 0; i < 8; ++i) { bf[i] = b0[i]; bf[8 + i] = b1[i]; }

            #pragma unroll
            for (int mt = 0; mt < 4; ++mt) {
                const __bf16* ap = &abuf[(mt * 16 + col) * APAD + kbl + koffA];
                const v8bf lo = *(const v8bf*)ap;
                const v8bf hi = *(const v8bf*)(ap + 16);
                v16bf af;
                #pragma unroll
                for (int i = 0; i < 8; ++i) { af[i] = lo[i]; af[8 + i] = hi[i]; }
                v8f* a = (mt == 0) ? &acc0 : (mt == 1) ? &acc1
                       : (mt == 2) ? &acc2 : &acc3;
                *a = __builtin_amdgcn_wmma_f32_16x16x32_bf16(
                         false, af, false, bf, (short)0, *a, false, false);
            }
        }
#if HAVE_TDM
        if (c + 1 < 8) {
            if (threadIdx.x < 32) tdm_wait();        // next chunk landed
            __syncthreads();
        }
#else
        __syncthreads();
#endif
    }

    const float bias = bx[g];
    #pragma unroll
    for (int mt = 0; mt < 4; ++mt) {
        const v8f* a = (mt == 0) ? &acc0 : (mt == 1) ? &acc1 : (mt == 2) ? &acc2 : &acc3;
        #pragma unroll
        for (int r = 0; r < 8; ++r) {
            const int m = r + hsel * 8;
            const int t = t0 + mt * 16 + m;
            XP[((size_t)t * B_ + b) * G4 + g] = (__bf16)((*a)[r] + bias);
        }
    }
}

// ---------------------------------------------------------------------------
// Phase 2: persistent recurrence, 64 WGs x 256 thr.
// WG = (mb: 16 batch rows, hc: 32 hidden cols). W_h slice TDM-staged in LDS
// (bf16, resident for all 1024 steps). h exchanged via 64KB L2-resident buf.
// ---------------------------------------------------------------------------
__global__ __launch_bounds__(256) void lstm_kernel(
    const float* __restrict__ masks, const __bf16* __restrict__ WhT,
    const __bf16* __restrict__ XP, __bf16* __restrict__ hbuf,
    unsigned* __restrict__ bar, float* __restrict__ out)
{
    extern __shared__ char smem_raw[];
    __bf16* Wt   = (__bf16*)smem_raw;                                // [128][WPAD]
    float*  actL = (float*)(smem_raw + 128 * WPAD * 2);              // [4][16][32]
    __bf16* hL   = (__bf16*)(smem_raw + 128 * WPAD * 2 + 4*16*32*4); // [16][HPAD]

    const int wg   = blockIdx.x;
    const int mb   = wg & 3;
    const int hc   = wg >> 2;
    const int tid  = threadIdx.x;
    const int lane = tid & 31;
    const int w    = tid >> 5;
    const int col  = lane & 15;
    const int hsel = lane >> 4;
    const int gate_g = w >> 1;
    const int jjb    = (w & 1) * 16;
    const int koffA  = hsel * 8;

#if HAVE_TDM
    if (tid < 32) {
        #pragma unroll
        for (int gt = 0; gt < 4; ++gt) {
            tdm_load_2d_bf16((unsigned)(uintptr_t)(Wt + gt * 32 * WPAD),
                             WhT + ((size_t)gt * 512 + hc * 32) * 512,
                             512, 2048, 512, /*tile*/512, 32, 7, 3);
        }
        tdm_wait();
    }
#else
    for (int idx = tid; idx < 128 * 512; idx += 256) {
        const int c = idx >> 9, k = idx & 511;
        const int g = (c >> 5) * 512 + hc * 32 + (c & 31);
        Wt[c * WPAD + k] = WhT[(size_t)g * 512 + k];
    }
#endif
    __syncthreads();

    float hreg[2] = {0.f, 0.f}, creg[2] = {0.f, 0.f};

    for (int t = 0; t < T_; ++t) {
        // stage h_prev rows (16 x 512 bf16) into padded LDS
        for (int ii = tid; ii < 16 * 64; ii += 256) {
            const int rr = ii >> 6, cc = (ii & 63) * 8;
            *(v8bf*)&hL[rr * HPAD + cc] =
                *(const v8bf*)&hbuf[(size_t)(mb * 16 + rr) * H_ + cc];
        }
        __syncthreads();

        const size_t xpbase =
            ((size_t)t * B_ + mb * 16) * G4 + gate_g * 512 + hc * 32 + jjb + col;
        if (t + 1 < T_)
            __builtin_prefetch(&XP[xpbase + (size_t)B_ * G4], 0, 1);

        // GEMM: wave w computes 16x16 act tile (n-tile w), K=512
        v8f acc = {0,0,0,0,0,0,0,0};
        #pragma unroll
        for (int kb = 0; kb < H_; kb += 32) {
            const __bf16* ha = &hL[col * HPAD + kb + koffA];
            const v8bf lo = *(const v8bf*)ha;
            const v8bf hi = *(const v8bf*)(ha + 16);
            v16bf af;
            #pragma unroll
            for (int i = 0; i < 8; ++i) { af[i] = lo[i]; af[8 + i] = hi[i]; }
            const __bf16* wp = &Wt[(w * 16 + col) * WPAD + kb + hsel * 8];
            const v8bf b0 = *(const v8bf*)wp;
            const v8bf b1 = *(const v8bf*)(wp + 16);
            v16bf bf;
            #pragma unroll
            for (int i = 0; i < 8; ++i) { bf[i] = b0[i]; bf[8 + i] = b1[i]; }
            acc = __builtin_amdgcn_wmma_f32_16x16x32_bf16(
                      false, af, false, bf, (short)0, acc, false, false);
        }

        // add x_proj, publish gate tile
        #pragma unroll
        for (int r = 0; r < 8; ++r) {
            const int m = r + hsel * 8;
            const float xv = (float)XP[xpbase + (size_t)m * G4];
            actL[(gate_g * 16 + m) * 32 + jjb + col] = acc[r] + xv;
        }
        __syncthreads();

        // elementwise gates: 2 cells per thread
        #pragma unroll
        for (int r = 0; r < 2; ++r) {
            const int e  = tid + 256 * r;
            const int bl = e >> 5, jj = e & 31;
            const float ai  = actL[(0 * 16 + bl) * 32 + jj];
            const float af_ = actL[(1 * 16 + bl) * 32 + jj];
            const float ac  = actL[(2 * 16 + bl) * 32 + jj];
            const float ao  = actL[(3 * 16 + bl) * 32 + jj];
            const float ig = 1.f / (1.f + __expf(-ai));
            const float fg = 1.f / (1.f + __expf(-af_));
            const float og = 1.f / (1.f + __expf(-ao));
            float cc = fg * creg[r] + ig * tanhf(ac);
            const float mk = masks[(size_t)(mb * 16 + bl) * T_ + t];
            cc = mk * cc + (1.f - mk) * creg[r];
            float hh = og * tanhf(cc);
            hh = mk * hh + (1.f - mk) * hreg[r];
            creg[r] = cc; hreg[r] = hh;

            const int bg = mb * 16 + bl;
            const int j  = hc * 32 + jj;
            const size_t o = ((size_t)bg * T_ + t) * H_ + j;
            out[o] = hh;
            out[(size_t)B_ * T_ * H_ + o] = cc;
            hbuf[(size_t)bg * H_ + j] = (__bf16)hh;
        }

        // device-wide barrier (monotonic, memset-reset each launch)
        __threadfence();
        __syncthreads();
        if (tid == 0) {
            __hip_atomic_fetch_add(bar, 1u, __ATOMIC_ACQ_REL, __HIP_MEMORY_SCOPE_AGENT);
            const unsigned target = 64u * (unsigned)(t + 1);
            while (__hip_atomic_load(bar, __ATOMIC_ACQUIRE, __HIP_MEMORY_SCOPE_AGENT) < target)
                __builtin_amdgcn_s_sleep(1);
        }
        __syncthreads();
    }
}

extern "C" void kernel_launch(void* const* d_in, const int* in_sizes, int n_in,
                              void* d_out, int out_size, void* d_ws, size_t ws_size,
                              hipStream_t stream) {
    const float* X  = (const float*)d_in[0];
    const float* Mk = (const float*)d_in[1];
    const float* Wx = (const float*)d_in[2];
    const float* bx = (const float*)d_in[3];
    const float* Wh = (const float*)d_in[4];
    float* out = (float*)d_out;

    // workspace layout
    unsigned* bar  = (unsigned*)d_ws;                                  // 256 B
    __bf16*  hbuf  = (__bf16*)((char*)d_ws + 256);                     // 64 KB
    __bf16*  WxT   = (__bf16*)((char*)d_ws + (1ull << 20));            // 2 MB
    __bf16*  WhT   = (__bf16*)((char*)d_ws + (4ull << 20));            // 2 MB
    __bf16*  Xb    = (__bf16*)((char*)d_ws + (8ull << 20));            // 64 MB
    __bf16*  XP    = (__bf16*)((char*)d_ws + (8ull << 20) + (64ull << 20)); // 256 MB

    hipMemsetAsync(d_ws, 0, 256 + (size_t)B_ * H_ * 2, stream);

    cvt_x_kernel<<<(B_ * T_ * NIN / 8) / 256, 256, 0, stream>>>(X, Xb);
    transpose_w_kernel<<<dim3((G4 * 512) / 256, 2), 256, 0, stream>>>(Wx, Wh, WxT, WhT);

    dim3 g1(G4 / 128, T_ / 64, B_);
    const size_t smem1 = (size_t)128 * WPAD * 2 + (size_t)2 * 64 * APAD * 2;
    xproj_kernel<<<g1, 256, smem1, stream>>>(Xb, WxT, bx, XP);

    const size_t smem2 = (size_t)128 * WPAD * 2 + 4 * 16 * 32 * 4 + (size_t)16 * HPAD * 2;
    lstm_kernel<<<64, 256, smem2, stream>>>(Mk, WhT, XP, hbuf, bar, out);
}
